// MultiHeadAttention_28089086116369
// MI455X (gfx1250) — matrix operand order repaired and checked by reading, not compiled, not run on hardware
//
#include <hip/hip_runtime.h>

// ---------------------------------------------------------------------------
// Feature detection: Tensor Data Mover builtin (arity differs per toolchain)
// ---------------------------------------------------------------------------
#if defined(__has_builtin)
#  if __has_builtin(__builtin_amdgcn_tensor_load_to_lds)
#    define HAVE_TDM 1
#  endif
#endif
#ifndef HAVE_TDM
#  define HAVE_TDM 0
#endif
#if __has_include(<hip/amd_detail/amd_gfx1250_TDM.h>)
#  define TDM_SIX_ARGS 1
#else
#  define TDM_SIX_ARGS 0
#endif

// ---------------------------------------------------------------------------
// Types for CDNA5 WMMA (wave32): 16x16x32 bf16 -> f32 accumulate
// ---------------------------------------------------------------------------
typedef __bf16 v16bf __attribute__((ext_vector_type(16)));
typedef float  v8f   __attribute__((ext_vector_type(8)));

__device__ __forceinline__ v8f wmma_bf16(v16bf a, v16bf b, v8f c) {
    // (neg_a, A, neg_b, B, c_mod, C, reuse_a, reuse_b)
    return __builtin_amdgcn_wmma_f32_16x16x32_bf16(false, a, false, b,
                                                   (short)0, c, false, false);
}

// bf16 round-to-nearest-even from f32 (bit trick)
__device__ __forceinline__ unsigned short f2bf(float f) {
    unsigned int u = __builtin_bit_cast(unsigned int, f);
    u += 0x7FFFu + ((u >> 16) & 1u);
    return (unsigned short)(u >> 16);
}

// ---------------------------------------------------------------------------
// DPP16 butterfly reductions within 16-lane halves (no LDS, pure VALU).
// quad_perm xor1 = 0xB1, xor2 = 0x4E, row_half_mirror = 0x141 (xor quads),
// row_mirror = 0x140 (xor octets).
// ---------------------------------------------------------------------------
template <int CTRL>
__device__ __forceinline__ float dpp_movf(float x) {
    int i = __builtin_bit_cast(int, x);
    int r = __builtin_amdgcn_update_dpp(i, i, CTRL, 0xF, 0xF, true);
    return __builtin_bit_cast(float, r);
}
__device__ __forceinline__ float reduce16_max(float x) {
    x = fmaxf(x, dpp_movf<0xB1>(x));
    x = fmaxf(x, dpp_movf<0x4E>(x));
    x = fmaxf(x, dpp_movf<0x141>(x));
    x = fmaxf(x, dpp_movf<0x140>(x));
    return x;
}
__device__ __forceinline__ float reduce16_sum(float x) {
    x += dpp_movf<0xB1>(x);
    x += dpp_movf<0x4E>(x);
    x += dpp_movf<0x141>(x);
    x += dpp_movf<0x140>(x);
    return x;
}

// Load a 16-element bf16 fragment from two 16-byte chunks (A-matrix layout).
__device__ __forceinline__ v16bf ld_frag2(const unsigned short* lo,
                                          const unsigned short* hi) {
    v16bf r;
    ((uint4*)&r)[0] = *(const uint4*)lo;
    ((uint4*)&r)[1] = *(const uint4*)hi;
    return r;
}
// Load a 16-element bf16 fragment from 32 contiguous bytes (B-matrix layout).
__device__ __forceinline__ v16bf ld_frag32(const unsigned short* p) {
    v16bf r;
    ((uint4*)&r)[0] = ((const uint4*)p)[0];
    ((uint4*)&r)[1] = ((const uint4*)p)[1];
    return r;
}

// ---------------------------------------------------------------------------
// TDM: 2D tile load global -> LDS (bf16 elements). D# per CDNA5 ISA §8.
// group0: [1:0]count=1 | [63:32]lds_addr | [120:64]global_addr | [127:126]type=2
// group1: [17:16]data_size=1(2B) | [79:48]tensor_dim0 | [111:80]tensor_dim1 |
//         [127:112]tile_dim0 | [143:128]tile_dim1 | [207:160]dim0_stride
// ---------------------------------------------------------------------------
#if HAVE_TDM
typedef unsigned int u32x4 __attribute__((ext_vector_type(4)));
typedef int          i32x8 __attribute__((ext_vector_type(8)));
typedef int          i32x4 __attribute__((ext_vector_type(4)));

__device__ __forceinline__ void tdm_load_2d_bf16(
    unsigned ldsAddr, const void* gptr,
    unsigned tile0, unsigned tile1,                 // elements (fast dim, rows)
    unsigned long long stride0,                     // row stride in elements
    unsigned tdim0, unsigned tdim1)                 // tensor extents (elements)
{
    const unsigned long long ga = (unsigned long long)gptr;
    u32x4 g0;
    g0[0] = 0x1u;                                            // count=1
    g0[1] = ldsAddr;                                         // lds_addr
    g0[2] = (unsigned)(ga & 0xFFFFFFFFu);                    // global_addr lo
    g0[3] = (unsigned)((ga >> 32) & 0x01FFFFFFu) | (2u << 30); // hi | type=2
    i32x8 g1;
    g1[0] = (int)(1u << 16);                                 // data_size=2B
    g1[1] = (int)((tdim0 & 0xFFFFu) << 16);                  // tensor_dim0 lo
    g1[2] = (int)(((tdim0 >> 16) & 0xFFFFu) | ((tdim1 & 0xFFFFu) << 16));
    g1[3] = (int)(((tdim1 >> 16) & 0xFFFFu) | ((tile0 & 0xFFFFu) << 16));
    g1[4] = (int)(tile1 & 0xFFFFu);                          // tile_dim1 (dim2=0)
    g1[5] = (int)(stride0 & 0xFFFFFFFFu);                    // dim0_stride lo
    g1[6] = (int)((stride0 >> 32) & 0xFFFFu);                // dim0_stride hi
    g1[7] = 0;
    i32x4 z4 = {0, 0, 0, 0};
#if TDM_SIX_ARGS
    i32x8 z8 = {0, 0, 0, 0, 0, 0, 0, 0};
    __builtin_amdgcn_tensor_load_to_lds(g0, g1, z4, z4, z8, 0);
#else
    __builtin_amdgcn_tensor_load_to_lds(g0, g1, z4, z4, 0);
#endif
}

__device__ __forceinline__ void wait_tensorcnt0() {
#if defined(__has_builtin) && __has_builtin(__builtin_amdgcn_s_wait_tensorcnt)
    __builtin_amdgcn_s_wait_tensorcnt(0);
#else
    asm volatile("s_wait_tensorcnt 0x0" ::: "memory");
#endif
}
#endif  // HAVE_TDM

// ---------------------------------------------------------------------------
// fp32 -> bf16 conversion (vectorized x4)
// ---------------------------------------------------------------------------
__global__ __launch_bounds__(256) void cvt_f32_bf16(const float* __restrict__ in,
                                                    unsigned short* __restrict__ out,
                                                    int n) {
    int i = (blockIdx.x * 256 + threadIdx.x) * 4;
    if (i < n) {
        float4 f = *(const float4*)(in + i);
        unsigned short o[4] = { f2bf(f.x), f2bf(f.y), f2bf(f.z), f2bf(f.w) };
        *(uint2*)(out + i) = *(const uint2*)o;
    }
}

// ---------------------------------------------------------------------------
// C[M,N] = A[M,K](bf16) @ B[N,K](bf16)^T + bias   (torch Linear: x @ W^T + b)
// WG = 128 threads = 4 waves (2x2); WG tile 128x128; wave tile 64x64:
// 16 WMMAs per 8 fragment loads per k-step.
// ---------------------------------------------------------------------------
template <bool F32OUT>
__global__ __launch_bounds__(128) void gemm_bf16(
    const unsigned short* __restrict__ A,   // [M,K] bf16
    const unsigned short* __restrict__ B,   // [N,K] bf16 (weight, row n = out col)
    const float* __restrict__ bias,         // [N]
    void* __restrict__ out,                 // [M,N] bf16 or f32
    int M, int N, int K)
{
    const int lane = threadIdx.x & 31;
    const int wv   = threadIdx.x >> 5;        // 0..3
    const int wm   = wv & 1;
    const int wn   = wv >> 1;
    const int mBase = blockIdx.y * 128 + wm * 64;
    const int nBase = blockIdx.x * 128 + wn * 64;
    const int lm = lane & 15;
    const int kg = lane >> 4;

    v8f acc[4][4];
#pragma unroll
    for (int i = 0; i < 4; ++i)
#pragma unroll
        for (int j = 0; j < 4; ++j) acc[i][j] = v8f{};

    for (int kk = 0; kk < K; kk += 32) {
        v16bf a[4], bf[4];
#pragma unroll
        for (int i = 0; i < 4; ++i) {
            const unsigned short* pa =
                A + (size_t)(mBase + i * 16 + lm) * K + kk + kg * 16;
            a[i] = ld_frag2(pa, pa + 8);
        }
#pragma unroll
        for (int j = 0; j < 4; ++j) {
            const unsigned short* pb =
                B + (size_t)(nBase + j * 16 + lm) * K + kk + kg * 16;
            bf[j] = ld_frag32(pb);
        }
#pragma unroll
        for (int i = 0; i < 4; ++i)
#pragma unroll
            for (int j = 0; j < 4; ++j)
                acc[i][j] = wmma_bf16(a[i], bf[j], acc[i][j]);
    }

#pragma unroll
    for (int i = 0; i < 4; ++i)
#pragma unroll
        for (int j = 0; j < 4; ++j) {
            const int col = nBase + j * 16 + lm;
            const float bv = bias[col];
#pragma unroll
            for (int r = 0; r < 8; ++r) {
                const int row = mBase + i * 16 + r + kg * 8;
                const float v = acc[i][j][r] + bv;
                if (F32OUT)
                    ((float*)out)[(size_t)row * N + col] = v;
                else
                    ((unsigned short*)out)[(size_t)row * N + col] = f2bf(v);
            }
        }
}

// ---------------------------------------------------------------------------
// Flash attention core (bf16 in/out), one WG per (b, h, 128 query rows).
// 8 waves x 16 query rows. 32-key blocks: K tile DMA'd by the Tensor Data
// Mover into LDS (TENSORcnt), V staged transposed by the WG, P through LDS.
// Online softmax over the full 32-key block with DPP16 butterflies.
// ---------------------------------------------------------------------------
__global__ __launch_bounds__(256) void flash_attn(
    const unsigned short* __restrict__ Q,   // [B*S, 512] bf16
    const unsigned short* __restrict__ Km,  // [B*S, 512] bf16
    const unsigned short* __restrict__ Vm,  // [B*S, 512] bf16
    unsigned short* __restrict__ O)         // [B*S, 512] bf16
{
    constexpr int Sseq = 2048, DM = 512, DK = 64;
    const int qb   = blockIdx.x;       // 0..15
    const int h    = blockIdx.y;       // 0..7
    const int b    = blockIdx.z;       // 0..3
    const int lane = threadIdx.x & 31;
    const int wv   = threadIdx.x >> 5; // 0..7
    const int lm   = lane & 15;
    const int kg   = lane >> 4;
    const int rowBase = b * Sseq + qb * 128 + wv * 16;
    const int hOff    = h * DK;

    __shared__ __align__(16) unsigned short ldsK[32 * 64];      // [key][d]
    __shared__ __align__(16) unsigned short ldsV[64 * 32];      // [d][key]
    __shared__ __align__(16) unsigned short ldsP[8][16 * 32];   // per-wave P

#if HAVE_TDM
    const unsigned ldsKaddr = (unsigned)(unsigned long long)(const void*)ldsK;
#endif

    // Preload Q fragments (16 rows x 64 d = two 16x32 A-frags).
    v16bf qf[2];
#pragma unroll
    for (int ks = 0; ks < 2; ++ks) {
        const unsigned short* pq =
            Q + (size_t)(rowBase + lm) * DM + hOff + ks * 32 + kg * 16;
        qf[ks] = ld_frag2(pq, pq + 8);
    }

    float mrow[8], lrow[8];
#pragma unroll
    for (int r = 0; r < 8; ++r) { mrow[r] = -3.0e38f; lrow[r] = 0.0f; }
    v8f o[4];
#pragma unroll
    for (int j = 0; j < 4; ++j) o[j] = v8f{};

    const float scale = 0.125f;  // 1/sqrt(64)

    for (int kb = 0; kb < Sseq / 32; ++kb) {
        // ---- stage K (TDM DMA) and transposed V (cooperative) ----
        {
            const int t   = threadIdx.x;
            const int key = t >> 3;        // 0..31
            const int d   = (t & 7) * 8;   // 0,8,...,56
            const size_t srow = (size_t)(b * Sseq + kb * 32 + key) * DM + hOff + d;
#if HAVE_TDM
            if (t < 32) {
                const unsigned short* gK =
                    Km + (size_t)(b * Sseq + kb * 32) * DM + hOff;
                // 2D tile: 64 elems/row (d, contiguous) x 32 rows (keys),
                // row stride 512 elements.
                tdm_load_2d_bf16(ldsKaddr, gK, 64, 32, 512ull, 64, 32);
            }
#else
            *(uint4*)&ldsK[key * 64 + d] = *(const uint4*)(Km + srow);
#endif
            uint4 vvec = *(const uint4*)(Vm + srow);
            const unsigned short* ve = (const unsigned short*)&vvec;
#pragma unroll
            for (int e = 0; e < 8; ++e) ldsV[(d + e) * 32 + key] = ve[e];
            if (kb + 1 < Sseq / 32) {
                __builtin_prefetch(Km + srow + 32 * DM, 0, 3);
                __builtin_prefetch(Vm + srow + 32 * DM, 0, 3);
            }
        }
#if HAVE_TDM
        wait_tensorcnt0();   // no-op for waves with TENSORcnt==0
#endif
        __syncthreads();

        // ---- both 16-key score tiles first (scheduler slack for WMMA) ----
        v8f s0 = v8f{}, s1 = v8f{};
#pragma unroll
        for (int ks = 0; ks < 2; ++ks) {
            v16bf kf0 = ld_frag32(&ldsK[lm * 64 + ks * 32 + kg * 16]);
            s0 = wmma_bf16(qf[ks], kf0, s0);
        }
#pragma unroll
        for (int ks = 0; ks < 2; ++ks) {
            v16bf kf1 = ld_frag32(&ldsK[(16 + lm) * 64 + ks * 32 + kg * 16]);
            s1 = wmma_bf16(qf[ks], kf1, s1);
        }

        // ---- one online-softmax update over the full 32-key block ----
#pragma unroll
        for (int r = 0; r < 8; ++r) {
            const float sv0 = s0[r] * scale;
            const float sv1 = s1[r] * scale;
            const float mx  = reduce16_max(fmaxf(sv0, sv1));
            const float mnew  = fmaxf(mrow[r], mx);
            const float alpha = __expf(mrow[r] - mnew);
            const float p0 = __expf(sv0 - mnew);
            const float p1 = __expf(sv1 - mnew);
            const float ps = reduce16_sum(p0 + p1);
            lrow[r] = lrow[r] * alpha + ps;
            mrow[r] = mnew;
#pragma unroll
            for (int j = 0; j < 4; ++j) o[j][r] *= alpha;
            // P in A-fragment layout: row = r + kg*8, cols lm and 16+lm
            ldsP[wv][(r + kg * 8) * 32 + lm]      = f2bf(p0);
            ldsP[wv][(r + kg * 8) * 32 + 16 + lm] = f2bf(p1);
        }

        // ---- O += P @ V (per-wave P tile; LDS in-order within wave) ----
        const unsigned short* pp = &ldsP[wv][lm * 32 + kg * 16];
        v16bf pf = ld_frag2(pp, pp + 8);
#pragma unroll
        for (int j = 0; j < 4; ++j) {
            v16bf vf = ld_frag32(&ldsV[(j * 16 + lm) * 32 + kg * 16]);
            o[j] = wmma_bf16(pf, vf, o[j]);
        }
        __syncthreads();
    }

    // ---- normalize and store ----
#pragma unroll
    for (int r = 0; r < 8; ++r) lrow[r] = 1.0f / lrow[r];
#pragma unroll
    for (int j = 0; j < 4; ++j)
#pragma unroll
        for (int r = 0; r < 8; ++r) {
            const float v = o[j][r] * lrow[r];
            const int row = rowBase + r + kg * 8;
            const int col = hOff + j * 16 + lm;
            O[(size_t)row * DM + col] = f2bf(v);
        }
}

// ---------------------------------------------------------------------------
// Host-side orchestration
// ---------------------------------------------------------------------------
extern "C" void kernel_launch(void* const* d_in, const int* in_sizes, int n_in,
                              void* d_out, int out_size, void* d_ws, size_t ws_size,
                              hipStream_t stream) {
    (void)in_sizes; (void)n_in; (void)out_size; (void)ws_size;

    const float* q   = (const float*)d_in[0];
    const float* k   = (const float*)d_in[1];
    const float* v   = (const float*)d_in[2];
    const float* W_q = (const float*)d_in[3];
    const float* b_q = (const float*)d_in[4];
    const float* W_k = (const float*)d_in[5];
    const float* b_k = (const float*)d_in[6];
    const float* W_v = (const float*)d_in[7];
    const float* b_v = (const float*)d_in[8];
    const float* W_o = (const float*)d_in[9];
    const float* b_o = (const float*)d_in[10];

    constexpr int B = 4, S = 2048, DM = 512;
    constexpr size_t ACT = (size_t)B * S * DM;   // 4,194,304 elements
    constexpr size_t WEI = (size_t)DM * DM;      //   262,144 elements

    unsigned short* ws = (unsigned short*)d_ws;
    unsigned short* xq = ws;
    unsigned short* xk = ws + ACT;
    unsigned short* xv = ws + 2 * ACT;
    unsigned short* Wqb = ws + 3 * ACT;
    unsigned short* Wkb = Wqb + WEI;
    unsigned short* Wvb = Wkb + WEI;
    unsigned short* Wob = Wvb + WEI;
    unsigned short* Qb  = Wob + WEI;
    unsigned short* Kb  = Qb + ACT;
    unsigned short* Vb  = Kb + ACT;
    unsigned short* At  = Vb + ACT;

    const int actGrid = (int)(ACT / 4 + 255) / 256;
    const int weiGrid = (int)(WEI / 4 + 255) / 256;
    cvt_f32_bf16<<<actGrid, 256, 0, stream>>>(q, xq, (int)ACT);
    cvt_f32_bf16<<<actGrid, 256, 0, stream>>>(k, xk, (int)ACT);
    cvt_f32_bf16<<<actGrid, 256, 0, stream>>>(v, xv, (int)ACT);
    cvt_f32_bf16<<<weiGrid, 256, 0, stream>>>(W_q, Wqb, (int)WEI);
    cvt_f32_bf16<<<weiGrid, 256, 0, stream>>>(W_k, Wkb, (int)WEI);
    cvt_f32_bf16<<<weiGrid, 256, 0, stream>>>(W_v, Wvb, (int)WEI);
    cvt_f32_bf16<<<weiGrid, 256, 0, stream>>>(W_o, Wob, (int)WEI);

    const dim3 gemmGrid(DM / 128, (B * S) / 128);  // (4, 64)
    gemm_bf16<false><<<gemmGrid, 128, 0, stream>>>(xq, Wqb, b_q, Qb, B * S, DM, DM);
    gemm_bf16<false><<<gemmGrid, 128, 0, stream>>>(xk, Wkb, b_k, Kb, B * S, DM, DM);
    gemm_bf16<false><<<gemmGrid, 128, 0, stream>>>(xv, Wvb, b_v, Vb, B * S, DM, DM);

    flash_attn<<<dim3(S / 128, 8, B), 256, 0, stream>>>(Qb, Kb, Vb, At);

    gemm_bf16<true><<<gemmGrid, 128, 0, stream>>>(At, Wob, b_o, d_out, B * S, DM, DM);
}
